// NNCRF_4217657884684
// MI455X (gfx1250) — matrix operand order repaired and checked by reading, not compiled, not run on hardware
//
#include <hip/hip_runtime.h>
#include <hip/hip_bf16.h>
#include <math.h>

// ---------------------------------------------------------------------------
// Types for CDNA5 WMMA (wave32): A/B are 16 bf16 per lane, C/D 8 f32 per lane.
// ---------------------------------------------------------------------------
typedef __attribute__((ext_vector_type(16))) __bf16 v16bf;
typedef __attribute__((ext_vector_type(8)))  __bf16 v8bf;
typedef __attribute__((ext_vector_type(8)))  float  v8f;

__device__ __forceinline__ v8f wmma_bf16f32(v16bf a, v16bf b, v8f c) {
  return __builtin_amdgcn_wmma_f32_16x16x32_bf16(false, a, false, b, (short)0, c,
                                                 false, false);
}
__device__ __forceinline__ v16bf cat8(v8bf lo, v8bf hi) {
  return __builtin_shufflevector(lo, hi, 0, 1, 2, 3, 4, 5, 6, 7,
                                 8, 9, 10, 11, 12, 13, 14, 15);
}
__device__ __forceinline__ float sigm(float x) { return 1.0f / (1.0f + expf(-x)); }
__device__ __forceinline__ int imin(int a, int b) { return a < b ? a : b; }

// ---------------------------------------------------------------------------
// Model dims
// ---------------------------------------------------------------------------
#define BB   64
#define TT   128
#define NN   (BB * TT)        // 8192
#define WDIM 300
#define DDIM 50
#define GDIM 300
#define HID  512
#define HD   256              // HID/2
#define LBL  17

// ---------------------------------------------------------------------------
// WMMA GEMM on pre-packed operands:
//   A: bf16 row-major [M][Kpad]   (Kpad = KT*32, zero padded)
//   B: bf16 fragment-packed [NT][KT][lane][16], NT = CDIV(N,64)*4, zero padded
//   C[M,N] = act( init + A@B ), init = Cin (may alias C) or bias or 0.
// One wave computes a 16x64 strip (4 accs share one A fragment per k-step).
// Inner loop is branchless: 2x16B A loads + 4x(32B B load + wmma).
// ---------------------------------------------------------------------------
__global__ void k_gemm_bf(const __bf16* __restrict__ A, int Kpad,
                          const __bf16* __restrict__ Bp, int KT,
                          float* __restrict__ C, int ldc,
                          const float* __restrict__ Cin, int ldcin,
                          const float* __restrict__ bias,
                          int M, int N, int relu) {
  const int lane = threadIdx.x & 31, wave = threadIdx.x >> 5;
  const int half = lane >> 4, l15 = lane & 15;
  const int tilesM  = (M + 15) >> 4;
  const int tilesN4 = (N + 63) >> 6;
  const int tid = blockIdx.x * 8 + wave;
  if (tid >= tilesM * tilesN4) return;          // wave-uniform
  const int m0  = (tid / tilesN4) * 16;
  const int n0  = (tid % tilesN4) * 64;
  const int nt0 = n0 >> 4;

  const int arow = imin(m0 + l15, M - 1);
  const v8bf*  __restrict__ Arow = (const v8bf*)(A + (size_t)arow * Kpad);
  const v16bf* __restrict__ Bf   = (const v16bf*)Bp;

  int ncol[4];
#pragma unroll
  for (int s = 0; s < 4; ++s) ncol[s] = imin(n0 + s * 16 + l15, N - 1);

  v8f acc[4];
#pragma unroll
  for (int s = 0; s < 4; ++s) {
#pragma unroll
    for (int r = 0; r < 8; ++r) {
      const int m = imin(m0 + r + half * 8, M - 1);
      float v = 0.0f;
      if (Cin)       v = Cin[(size_t)m * ldcin + ncol[s]];
      else if (bias) v = bias[ncol[s]];
      acc[s][r] = v;
    }
  }

  for (int ks = 0; ks < KT; ++ks) {
    const v16bf af = cat8(Arow[ks * 4 + 2 * half], Arow[ks * 4 + 2 * half + 1]);
#pragma unroll
    for (int s = 0; s < 4; ++s) {
      const v16bf bfr = Bf[((size_t)(nt0 + s) * KT + ks) * 32 + lane];
      acc[s] = wmma_bf16f32(af, bfr, acc[s]);
    }
  }

#pragma unroll
  for (int s = 0; s < 4; ++s) {
#pragma unroll
    for (int r = 0; r < 8; ++r) {
      const int m = m0 + r + half * 8, n = n0 + s * 16 + l15;
      if (m < M && n < N) {
        float v = acc[s][r];
        if (relu) v = fmaxf(v, 0.0f);
        C[(size_t)m * ldc + n] = v;
      }
    }
  }
}

// ---------------------------------------------------------------------------
// Operand packers.
// k_packB: dst[nt][kt][lane][e] = (bf16) B[k,n] (zero padded)
//          k = kt*32 + e + (lane>>4)*16, n = nt*16 + (lane&15)
// k_packA: f32 dense [M][K] -> bf16 row-major [M][Kpad], zero padded.
// ---------------------------------------------------------------------------
__global__ void k_packB(const float* __restrict__ B, int ldb, int K, int N,
                        __bf16* __restrict__ d, int KT, int NT) {
  int i = blockIdx.x * 256 + threadIdx.x;
  if (i >= NT * KT * 512) return;
  const int e = i & 15, lane = (i >> 4) & 31;
  const int kt = (i >> 9) % KT, nt = i / (KT * 512);
  const int k = kt * 32 + e + (lane >> 4) * 16;
  const int n = nt * 16 + (lane & 15);
  d[i] = (k < K && n < N) ? (__bf16)B[(size_t)k * ldb + n] : (__bf16)0.0f;
}
__global__ void k_packA(const float* __restrict__ A, int M, int K, int Kpad,
                        __bf16* __restrict__ d) {
  int i = blockIdx.x * 256 + threadIdx.x;
  if (i >= M * Kpad) return;
  const int r = i / Kpad, k = i - r * Kpad;
  d[i] = (k < K) ? (__bf16)A[(size_t)r * K + k] : (__bf16)0.0f;
}

// ---------------------------------------------------------------------------
// Persistent _mylstm recurrence. 1 block, 1024 threads = 32 waves.
// 128 (b,j) tiles, 4/wave. Per k-step one A fragment (2 x ds b128 from the
// LDS bf16 h double-buffer) feeds 6 gate WMMAs; B operands are contiguous
// 32B pre-packed fragments. c stays in VGPRs for all 128 steps.
// ---------------------------------------------------------------------------
__global__ void k_mylstm(const float* __restrict__ X1, const float* __restrict__ X2,
                         const float* __restrict__ X3, const float* __restrict__ X4,
                         const float* __restrict__ X11, const float* __restrict__ X44,
                         const __bf16* __restrict__ whcat, float* __restrict__ out) {
  extern __shared__ char smem[];
  __bf16* h0 = (__bf16*)smem;        // 64*512 bf16
  __bf16* h1 = h0 + BB * HID;
  const int lane = threadIdx.x & 31, wave = threadIdx.x >> 5;
  const int half = lane >> 4, l15 = lane & 15;
  const v16bf* __restrict__ wfrag = (const v16bf*)whcat;

  for (int i = threadIdx.x; i < BB * HID; i += blockDim.x) h0[i] = (__bf16)0.0f;
  __syncthreads();

  const float* Xs[6] = {X1, X2, X3, X4, X11, X44};
  v8f cst[4];
#pragma unroll
  for (int p = 0; p < 4; ++p)
#pragma unroll
    for (int r = 0; r < 8; ++r) cst[p][r] = 0.0f;

  for (int t = 0; t < TT; ++t) {
    __bf16* hc = (t & 1) ? h1 : h0;
    __bf16* hn = (t & 1) ? h0 : h1;
#pragma unroll
    for (int p = 0; p < 4; ++p) {
      const int pid = wave * 4 + p;          // 0..127
      const int tm = pid >> 5, tn = pid & 31;
      const int m0 = tm * 16, n0 = tn * 16;
      const v8bf* __restrict__ hrow = (const v8bf*)(hc + (m0 + l15) * HID);

      v8f ga[6];
#pragma unroll
      for (int g = 0; g < 6; ++g)
#pragma unroll
        for (int r = 0; r < 8; ++r) {
          const int b = m0 + r + half * 8;
          ga[g][r] = Xs[g][((size_t)b * TT + t) * HID + n0 + l15];
        }

      for (int ks = 0; ks < HID / 32; ++ks) {
        const v16bf af = cat8(hrow[ks * 4 + 2 * half], hrow[ks * 4 + 2 * half + 1]);
#pragma unroll
        for (int g = 0; g < 6; ++g) {
          const v16bf bfr = wfrag[((g * 32 + tn) * (HID / 32) + ks) * 32 + lane];
          ga[g] = wmma_bf16f32(af, bfr, ga[g]);
        }
      }

#pragma unroll
      for (int r = 0; r < 8; ++r) {
        const float iv = sigm(ga[0][r]), ov = sigm(ga[1][r]), fv = sigm(ga[2][r]);
        const float uv = tanhf(ga[3][r]), i2 = sigm(ga[4][r]), u2 = tanhf(ga[5][r]);
        const float cv = iv * uv + i2 * u2 + fv * cst[p][r];
        cst[p][r] = cv;
        const float hv = ov * tanhf(cv);
        const int b = m0 + r + half * 8, j = n0 + l15;
        hn[b * HID + j] = (__bf16)hv;
        out[((size_t)b * TT + t) * HID + j] = hv;
      }
    }
    __syncthreads();
  }
}

// ---------------------------------------------------------------------------
// Persistent BiLSTM layer. grid=2 (block 0 = fwd, 1 = bwd), 1024 threads.
// Each wave owns 2 (b,jh) patches; one A fragment feeds 4 gate WMMAs.
// ---------------------------------------------------------------------------
__global__ void k_bilstm(const float* __restrict__ xihF, const float* __restrict__ xihB,
                         const __bf16* __restrict__ whhF, const __bf16* __restrict__ whhB,
                         float* __restrict__ out) {
  extern __shared__ char smem[];
  __bf16* hb0 = (__bf16*)smem;       // 64*256 bf16
  __bf16* hb1 = hb0 + BB * HD;
  const int dir = blockIdx.x;
  const float* __restrict__ xih   = dir ? xihB : xihF;
  const v16bf* __restrict__ wfrag = (const v16bf*)(dir ? whhB : whhF);
  const int lane = threadIdx.x & 31, wave = threadIdx.x >> 5;
  const int half = lane >> 4, l15 = lane & 15;

  for (int i = threadIdx.x; i < BB * HD; i += blockDim.x) hb0[i] = (__bf16)0.0f;
  __syncthreads();

  v8f cst[2];
#pragma unroll
  for (int p = 0; p < 2; ++p)
#pragma unroll
    for (int r = 0; r < 8; ++r) cst[p][r] = 0.0f;

  for (int t = 0; t < TT; ++t) {
    const int tt = dir ? (TT - 1 - t) : t;
    __bf16* hc = (t & 1) ? hb1 : hb0;
    __bf16* hn = (t & 1) ? hb0 : hb1;
#pragma unroll
    for (int p = 0; p < 2; ++p) {
      const int pid = wave * 2 + p;        // 0..63
      const int tm = pid >> 4, tn = pid & 15;
      const int m0 = tm * 16, n0 = tn * 16;
      const v8bf* __restrict__ hrow = (const v8bf*)(hc + (m0 + l15) * HD);

      v8f ga[4];
#pragma unroll
      for (int g = 0; g < 4; ++g)
#pragma unroll
        for (int r = 0; r < 8; ++r) {
          const int b = m0 + r + half * 8;
          ga[g][r] = xih[((size_t)b * TT + tt) * (4 * HD) + g * HD + n0 + l15];
        }

      for (int ks = 0; ks < HD / 32; ++ks) {
        const v16bf af = cat8(hrow[ks * 4 + 2 * half], hrow[ks * 4 + 2 * half + 1]);
#pragma unroll
        for (int g = 0; g < 4; ++g) {
          const v16bf bfr = wfrag[((g * 16 + tn) * (HD / 32) + ks) * 32 + lane];
          ga[g] = wmma_bf16f32(af, bfr, ga[g]);
        }
      }

#pragma unroll
      for (int r = 0; r < 8; ++r) {
        const float iv = sigm(ga[0][r]), fv = sigm(ga[1][r]);
        const float gv = tanhf(ga[2][r]), ov = sigm(ga[3][r]);
        const float cv = fv * cst[p][r] + iv * gv;
        cst[p][r] = cv;
        const float hv = ov * tanhf(cv);
        const int b = m0 + r + half * 8, j = n0 + l15;
        hn[b * HD + j] = (__bf16)hv;
        out[((size_t)b * TT + tt) * HID + dir * HD + j] = hv;
      }
    }
    __syncthreads();
  }
}

// ---------------------------------------------------------------------------
// Recurrent-weight fragment packers (same fragment layout as k_packB).
// ---------------------------------------------------------------------------
__global__ void k_pack6(const float* W1, const float* W2, const float* W3,
                        const float* W4, const float* W11, const float* W44,
                        __bf16* __restrict__ d) {
  int i = blockIdx.x * 256 + threadIdx.x;
  if (i >= 6 * HID * HID) return;
  const int e = i & 15, lane = (i >> 4) & 31, ks = (i >> 9) & 15;
  const int tn = (i >> 13) & 31, g = i >> 18;
  const int k = ks * 32 + e + (lane >> 4) * 16;
  const int n = tn * 16 + (lane & 15);
  const float* Ws[6] = {W1, W2, W3, W4, W11, W44};
  d[i] = (__bf16)Ws[g][(size_t)k * HID + n];   // rows 0..HID-1 = h-part
}
__global__ void k_pack2(const float* __restrict__ F, const float* __restrict__ Bw,
                        __bf16* __restrict__ d) {
  int i = blockIdx.x * 256 + threadIdx.x;
  if (i >= 2 * HD * 4 * HD) return;
  const int dir = i >> 18, rem = i & ((1 << 18) - 1);
  const int e = rem & 15, lane = (rem >> 4) & 31, ks = (rem >> 9) & 7;
  const int tn = (rem >> 12) & 15, g = rem >> 16;
  const int k = ks * 32 + e + (lane >> 4) * 16;
  const int n = tn * 16 + (lane & 15);
  const float* Wsrc = dir ? Bw : F;
  d[i] = (__bf16)Wsrc[(size_t)k * (4 * HD) + g * HD + n];
}

// ---------------------------------------------------------------------------
// Small helper kernels
// ---------------------------------------------------------------------------
__global__ void k_zero(float* p, int n) {
  int i = blockIdx.x * 256 + threadIdx.x;
  if (i < n) p[i] = 0.0f;
}
__global__ void k_add(const float* a, const float* b, float* o, int n) {
  int i = blockIdx.x * 256 + threadIdx.x;
  if (i < n) o[i] = a[i] + b[i];
}
__global__ void k_gather_we(const int* __restrict__ wi, const float* __restrict__ tab,
                            float* __restrict__ we) {
  int i = blockIdx.x * 256 + threadIdx.x;
  if (i >= NN * WDIM) return;
  int r = i / WDIM, c = i - r * WDIM;
  we[i] = tab[(size_t)wi[r] * WDIM + c];
}
__global__ void k_scat_dep(const int* __restrict__ lab, const int* __restrict__ ei,
                           const float* __restrict__ dtab, float* __restrict__ dep) {
  int i = blockIdx.x * 256 + threadIdx.x;
  if (i >= NN * DDIM) return;
  int e = i / DDIM, c = i - e * DDIM;
  int col = ei[NN + e];
  atomicAdd(&dep[(size_t)col * DDIM + c], dtab[(size_t)lab[e] * DDIM + c]);
}
__global__ void k_deg(const int* __restrict__ ei, float* __restrict__ deg) {
  int e = blockIdx.x * 256 + threadIdx.x;
  if (e >= NN) return;
  atomicAdd(&deg[ei[NN + e]], 1.0f);
}
__global__ void k_dinv(const float* __restrict__ deg, float* __restrict__ dinv) {
  int i = blockIdx.x * 256 + threadIdx.x;
  if (i >= NN) return;
  dinv[i] = rsqrtf(fmaxf(deg[i] + 1.0f, 1e-12f));   // +1 self loop
}
__global__ void k_scat_gcn(const int* __restrict__ ei, const float* __restrict__ dinv,
                           const float* __restrict__ xw, float* __restrict__ gcn) {
  int i = blockIdx.x * 256 + threadIdx.x;
  if (i >= NN * GDIM) return;
  int e = i / GDIM, c = i - e * GDIM;
  int row = ei[e], col = ei[NN + e];
  atomicAdd(&gcn[(size_t)col * GDIM + c],
            dinv[row] * dinv[col] * xw[(size_t)row * GDIM + c]);
}
__global__ void k_gcn_fin(const float* __restrict__ dinv, const float* __restrict__ xw,
                          const float* __restrict__ b, float* __restrict__ gcn) {
  int i = blockIdx.x * 256 + threadIdx.x;
  if (i >= NN * GDIM) return;
  int r = i / GDIM, c = i - r * GDIM;
  gcn[i] = fmaxf(gcn[i] + dinv[r] * dinv[r] * xw[i] + b[c], 0.0f);
}
__global__ void k_ln_relu(float* __restrict__ X, const float* __restrict__ g,
                          const float* __restrict__ bt, int rows, int ncol) {
  int r = blockIdx.x * blockDim.x + threadIdx.x;
  if (r >= rows) return;
  float* x = X + (size_t)r * ncol;
  float mu = 0.0f;
  for (int j = 0; j < ncol; ++j) mu += x[j];
  mu /= ncol;
  float var = 0.0f;
  for (int j = 0; j < ncol; ++j) { float d = x[j] - mu; var += d * d; }
  var /= ncol;
  const float s = rsqrtf(var + 1e-5f);
  for (int j = 0; j < ncol; ++j) x[j] = fmaxf((x[j] - mu) * s * g[j] + bt[j], 0.0f);
}
__global__ void k_pool(const float* __restrict__ o1, float* __restrict__ pooled) {
  const int b = blockIdx.x, j = threadIdx.x;   // block 512
  float s = 0.0f, mx = -INFINITY;
  for (int t = 0; t < TT; ++t) {
    float v = o1[((size_t)b * TT + t) * HID + j];
    s += v; mx = fmaxf(mx, v);
  }
  pooled[b * 3 * HID + j]           = s * (1.0f / TT);
  pooled[b * 3 * HID + HID + j]     = mx;
  pooled[b * 3 * HID + 2 * HID + j] = o1[((size_t)b * TT + TT - 1) * HID + j];
}

// ---------------------------------------------------------------------------
// Host side
// ---------------------------------------------------------------------------
#define CDIV(a, b) (((a) + (b)-1) / (b))

static void packB(hipStream_t s, const float* B, int ldb, int K, int N, __bf16* d) {
  const int KT = CDIV(K, 32), NT = CDIV(N, 64) * 4;
  k_packB<<<CDIV(NT * KT * 512, 256), 256, 0, s>>>(B, ldb, K, N, d, KT, NT);
}
static void packA(hipStream_t s, const float* A, int M, int K, __bf16* d) {
  const int Kpad = CDIV(K, 32) * 32;
  k_packA<<<CDIV(M * Kpad, 256), 256, 0, s>>>(A, M, K, Kpad, d);
}
static void gemmbf(hipStream_t s, const __bf16* A, int K, const __bf16* Bp,
                   float* C, int ldc, const float* Cin, int ldcin,
                   const float* bias, int M, int N, int relu) {
  const int KT = CDIV(K, 32), Kpad = KT * 32;
  const int tiles = CDIV(M, 16) * CDIV(N, 64);
  k_gemm_bf<<<CDIV(tiles, 8), 256, 0, s>>>(A, Kpad, Bp, KT, C, ldc, Cin, ldcin,
                                           bias, M, N, relu);
}

enum {
  I_WORD_INPUTS = 0, I_DEP_LABELS, I_EDGE_INDEX, I_WORD_EMB, I_DEP_EMB,
  I_GCN_W, I_GCN_B,
  I_W1, I_B1, I_W2, I_B2, I_W3, I_B3, I_W4, I_B4, I_W11, I_B11, I_W44, I_B44,
  I_L0F_WIH, I_L0F_WHH, I_L0F_BIH, I_L0F_BHH,
  I_L0B_WIH, I_L0B_WHH, I_L0B_BIH, I_L0B_BHH,
  I_L1F_WIH, I_L1F_WHH, I_L1F_BIH, I_L1F_BHH,
  I_L1B_WIH, I_L1B_WHH, I_L1B_BIH, I_L1B_BHH,
  I_NER_W1, I_NER_B1, I_NER_G1, I_NER_BT1, I_NER_W2, I_NER_B2,
  I_SA_W1, I_SA_B1, I_SA_G1, I_SA_BT1, I_SA_W2, I_SA_B2
};

extern "C" void kernel_launch(void* const* d_in, const int* in_sizes, int n_in,
                              void* d_out, int out_size, void* d_ws, size_t ws_size,
                              hipStream_t stream) {
  (void)in_sizes; (void)n_in; (void)out_size; (void)ws_size;
  const int*   wi   = (const int*)d_in[I_WORD_INPUTS];
  const int*   lab  = (const int*)d_in[I_DEP_LABELS];
  const int*   ei   = (const int*)d_in[I_EDGE_INDEX];
  const float* wtab = (const float*)d_in[I_WORD_EMB];
  const float* dtab = (const float*)d_in[I_DEP_EMB];
  const float* Wg   = (const float*)d_in[I_GCN_W];
  const float* bg   = (const float*)d_in[I_GCN_B];
  const float* W1   = (const float*)d_in[I_W1];   const float* b1  = (const float*)d_in[I_B1];
  const float* W2   = (const float*)d_in[I_W2];   const float* b2  = (const float*)d_in[I_B2];
  const float* W3   = (const float*)d_in[I_W3];   const float* b3  = (const float*)d_in[I_B3];
  const float* W4   = (const float*)d_in[I_W4];   const float* b4  = (const float*)d_in[I_B4];
  const float* W11  = (const float*)d_in[I_W11];  const float* b11 = (const float*)d_in[I_B11];
  const float* W44  = (const float*)d_in[I_W44];  const float* b44 = (const float*)d_in[I_B44];
  float* out = (float*)d_out;   // [0,139264) = ner (8192x17), then sa (64x2)

  // ---- workspace: f32 region (sizes multiple of 8), then bf16 region ----
  float* W = (float*)d_ws;
  size_t off = 0;
  auto alloc = [&](size_t n) { float* p = W + off; off += n; return p; };
  float* we    = alloc((size_t)NN * WDIM);
  float* dep   = alloc((size_t)NN * DDIM);
  float* deg   = alloc(NN);
  float* dinv  = alloc(NN);
  float* xw    = alloc((size_t)NN * GDIM);
  float* gcn   = alloc((size_t)NN * GDIM);
  float* Rbig  = alloc((size_t)6 * NN * HID);  // X1..X44; earlier xih/out0
  float* out1  = alloc((size_t)NN * HID);
  float* lstmo = alloc((size_t)NN * HID);
  float* nerh  = alloc((size_t)NN * HD);
  float* pooled= alloc((size_t)BB * 3 * HID);
  float* sah   = alloc((size_t)BB * 64);
  float* bsum  = alloc(4 * 1024);

  __bf16* bbase = (__bf16*)(W + off);
  size_t boff = 0;
  auto balloc = [&](size_t n) { __bf16* p = bbase + boff; boff += (n + 15) & ~(size_t)15; return p; };
  __bf16* whcat  = balloc((size_t)6 * HID * HID);
  __bf16* whhpk  = balloc((size_t)2 * HD * 4 * HD);
  // activation bf16 copies (row-major, K padded to mult of 32)
  __bf16* webf    = balloc((size_t)NN * 320);
  __bf16* depbf   = balloc((size_t)NN * 64);
  __bf16* gcnbf   = balloc((size_t)NN * 320);
  __bf16* out0bf  = balloc((size_t)NN * HID);
  __bf16* lstmobf = balloc((size_t)NN * HID);
  __bf16* nerhbf  = balloc((size_t)NN * HD);
  __bf16* pooledbf= balloc((size_t)BB * 3 * HID);
  __bf16* sahbf   = balloc((size_t)BB * 64);
  // fragment-packed weights
  __bf16* Bwg_x = balloc((size_t)20 * 10 * 512);
  __bf16* Bwg_d = balloc((size_t)20 * 2 * 512);
  __bf16* Bsyn[12];
  for (int i = 0; i < 12; ++i) Bsyn[i] = balloc((size_t)32 * 10 * 512);
  __bf16* Bih0f = balloc((size_t)64 * 10 * 512);
  __bf16* Bih0b = balloc((size_t)64 * 10 * 512);
  __bf16* Bih1f = balloc((size_t)64 * 16 * 512);
  __bf16* Bih1b = balloc((size_t)64 * 16 * 512);
  __bf16* Bnw1  = balloc((size_t)16 * 16 * 512);
  __bf16* Bnw2  = balloc((size_t)4 * 8 * 512);
  __bf16* Bsw1  = balloc((size_t)4 * 48 * 512);
  __bf16* Bsw2  = balloc((size_t)4 * 2 * 512);

  float* xihF = Rbig;                                      // 8192x1024
  float* xihB = Rbig + (size_t)NN * 4 * HD;                // 8192x1024
  float* out0 = Rbig + (size_t)2 * NN * 4 * HD;            // 8192x512
  float* X[6];
  for (int g = 0; g < 6; ++g) X[g] = Rbig + (size_t)g * NN * HID;

  // ---- pack all weights up front (independent of activations) ----
  packB(stream, Wg, GDIM, WDIM, GDIM, Bwg_x);
  packB(stream, Wg + (size_t)WDIM * GDIM, GDIM, DDIM, GDIM, Bwg_d);
  // mylstm X blocks: [512:812)=we, [812:1112)=gcn(xt), [1112:1412)=gcn(mt)
  packB(stream, W1 + (size_t)512 * HID, HID, WDIM, HID, Bsyn[0]);   // W1 we
  packB(stream, W1 + (size_t)812 * HID, HID, GDIM, HID, Bsyn[1]);   // W1 gcn
  packB(stream, W2 + (size_t)512 * HID, HID, WDIM, HID, Bsyn[2]);
  packB(stream, W2 + (size_t)812 * HID, HID, GDIM, HID, Bsyn[3]);
  packB(stream, W2 + (size_t)1112 * HID, HID, GDIM, HID, Bsyn[4]);
  packB(stream, W3 + (size_t)512 * HID, HID, WDIM, HID, Bsyn[5]);
  packB(stream, W3 + (size_t)812 * HID, HID, GDIM, HID, Bsyn[6]);
  packB(stream, W3 + (size_t)1112 * HID, HID, GDIM, HID, Bsyn[7]);
  packB(stream, W4 + (size_t)512 * HID, HID, WDIM, HID, Bsyn[8]);
  packB(stream, W4 + (size_t)812 * HID, HID, GDIM, HID, Bsyn[9]);
  packB(stream, W11 + (size_t)512 * HID, HID, GDIM, HID, Bsyn[10]);
  packB(stream, W44 + (size_t)512 * HID, HID, GDIM, HID, Bsyn[11]);
  packB(stream, (const float*)d_in[I_L0F_WIH], 4 * HD, WDIM, 4 * HD, Bih0f);
  packB(stream, (const float*)d_in[I_L0B_WIH], 4 * HD, WDIM, 4 * HD, Bih0b);
  packB(stream, (const float*)d_in[I_L1F_WIH], 4 * HD, HID, 4 * HD, Bih1f);
  packB(stream, (const float*)d_in[I_L1B_WIH], 4 * HD, HID, 4 * HD, Bih1b);
  packB(stream, (const float*)d_in[I_NER_W1], HD, HID, HD, Bnw1);
  packB(stream, (const float*)d_in[I_NER_W2], LBL, HD, LBL, Bnw2);
  packB(stream, (const float*)d_in[I_SA_W1], 64, 3 * HID, 64, Bsw1);
  packB(stream, (const float*)d_in[I_SA_W2], 2, 64, 2, Bsw2);
  k_pack6<<<CDIV(6 * HID * HID, 256), 256, 0, stream>>>(W1, W2, W3, W4, W11, W44, whcat);

  // ---- word embedding gather ----
  k_gather_we<<<CDIV(NN * WDIM, 256), 256, 0, stream>>>(wi, wtab, we);
  packA(stream, we, NN, WDIM, webf);

  // ================= SA path (BiLSTM) =================
  k_add<<<4, 256, 0, stream>>>((const float*)d_in[I_L0F_BIH], (const float*)d_in[I_L0F_BHH], bsum + 0,    1024);
  k_add<<<4, 256, 0, stream>>>((const float*)d_in[I_L0B_BIH], (const float*)d_in[I_L0B_BHH], bsum + 1024, 1024);
  k_add<<<4, 256, 0, stream>>>((const float*)d_in[I_L1F_BIH], (const float*)d_in[I_L1F_BHH], bsum + 2048, 1024);
  k_add<<<4, 256, 0, stream>>>((const float*)d_in[I_L1B_BIH], (const float*)d_in[I_L1B_BHH], bsum + 3072, 1024);

  k_pack2<<<CDIV(2 * HD * 4 * HD, 256), 256, 0, stream>>>(
      (const float*)d_in[I_L0F_WHH], (const float*)d_in[I_L0B_WHH], whhpk);
  gemmbf(stream, webf, WDIM, Bih0f, xihF, 4 * HD, nullptr, 0, bsum + 0,    NN, 4 * HD, 0);
  gemmbf(stream, webf, WDIM, Bih0b, xihB, 4 * HD, nullptr, 0, bsum + 1024, NN, 4 * HD, 0);
  k_bilstm<<<2, 1024, 2 * BB * HD * sizeof(__bf16), stream>>>(
      xihF, xihB, whhpk, whhpk + (size_t)HD * 4 * HD, out0);
  packA(stream, out0, NN, HID, out0bf);

  k_pack2<<<CDIV(2 * HD * 4 * HD, 256), 256, 0, stream>>>(
      (const float*)d_in[I_L1F_WHH], (const float*)d_in[I_L1B_WHH], whhpk);
  gemmbf(stream, out0bf, HID, Bih1f, xihF, 4 * HD, nullptr, 0, bsum + 2048, NN, 4 * HD, 0);
  gemmbf(stream, out0bf, HID, Bih1b, xihB, 4 * HD, nullptr, 0, bsum + 3072, NN, 4 * HD, 0);
  k_bilstm<<<2, 1024, 2 * BB * HD * sizeof(__bf16), stream>>>(
      xihF, xihB, whhpk, whhpk + (size_t)HD * 4 * HD, out1);

  k_pool<<<BB, HID, 0, stream>>>(out1, pooled);
  packA(stream, pooled, BB, 3 * HID, pooledbf);
  gemmbf(stream, pooledbf, 3 * HID, Bsw1, sah, 64, nullptr, 0,
         (const float*)d_in[I_SA_B1], BB, 64, 0);
  k_ln_relu<<<1, 64, 0, stream>>>(sah, (const float*)d_in[I_SA_G1],
                                  (const float*)d_in[I_SA_BT1], BB, 64);
  packA(stream, sah, BB, 64, sahbf);
  gemmbf(stream, sahbf, 64, Bsw2, out + NN * LBL, 2, nullptr, 0,
         (const float*)d_in[I_SA_B2], BB, 2, 0);

  // ================= GCN + mylstm + NER path =================
  k_zero<<<CDIV(NN * DDIM, 256), 256, 0, stream>>>(dep, NN * DDIM);
  k_zero<<<CDIV(NN, 256), 256, 0, stream>>>(deg, NN);
  k_zero<<<CDIV(NN * GDIM, 256), 256, 0, stream>>>(gcn, NN * GDIM);
  k_scat_dep<<<CDIV(NN * DDIM, 256), 256, 0, stream>>>(lab, ei, dtab, dep);
  packA(stream, dep, NN, DDIM, depbf);
  k_deg<<<CDIV(NN, 256), 256, 0, stream>>>(ei, deg);
  k_dinv<<<CDIV(NN, 256), 256, 0, stream>>>(deg, dinv);

  gemmbf(stream, webf, WDIM, Bwg_x, xw, GDIM, nullptr, 0, nullptr, NN, GDIM, 0);
  gemmbf(stream, depbf, DDIM, Bwg_d, xw, GDIM, xw, GDIM, nullptr, NN, GDIM, 0);
  k_scat_gcn<<<CDIV(NN * GDIM, 256), 256, 0, stream>>>(ei, dinv, xw, gcn);
  k_gcn_fin<<<CDIV(NN * GDIM, 256), 256, 0, stream>>>(dinv, xw, bg, gcn);
  packA(stream, gcn, NN, GDIM, gcnbf);

  gemmbf(stream, webf,  WDIM, Bsyn[0], X[0], HID, nullptr, 0, b1,  NN, HID, 0);
  gemmbf(stream, gcnbf, GDIM, Bsyn[1], X[0], HID, X[0], HID, nullptr, NN, HID, 0);
  gemmbf(stream, webf,  WDIM, Bsyn[2], X[1], HID, nullptr, 0, b2,  NN, HID, 0);
  gemmbf(stream, gcnbf, GDIM, Bsyn[3], X[1], HID, X[1], HID, nullptr, NN, HID, 0);
  gemmbf(stream, gcnbf, GDIM, Bsyn[4], X[1], HID, X[1], HID, nullptr, NN, HID, 0);
  gemmbf(stream, webf,  WDIM, Bsyn[5], X[2], HID, nullptr, 0, b3,  NN, HID, 0);
  gemmbf(stream, gcnbf, GDIM, Bsyn[6], X[2], HID, X[2], HID, nullptr, NN, HID, 0);
  gemmbf(stream, gcnbf, GDIM, Bsyn[7], X[2], HID, X[2], HID, nullptr, NN, HID, 0);
  gemmbf(stream, webf,  WDIM, Bsyn[8], X[3], HID, nullptr, 0, b4,  NN, HID, 0);
  gemmbf(stream, gcnbf, GDIM, Bsyn[9], X[3], HID, X[3], HID, nullptr, NN, HID, 0);
  gemmbf(stream, gcnbf, GDIM, Bsyn[10], X[4], HID, nullptr, 0, b11, NN, HID, 0);
  gemmbf(stream, gcnbf, GDIM, Bsyn[11], X[5], HID, nullptr, 0, b44, NN, HID, 0);

  k_mylstm<<<1, 1024, 2 * BB * HID * sizeof(__bf16), stream>>>(
      X[0], X[1], X[2], X[3], X[4], X[5], whcat, lstmo);
  packA(stream, lstmo, NN, HID, lstmobf);

  // NER head
  gemmbf(stream, lstmobf, HID, Bnw1, nerh, HD, nullptr, 0,
         (const float*)d_in[I_NER_B1], NN, HD, 0);
  k_ln_relu<<<CDIV(NN, 128), 128, 0, stream>>>(nerh, (const float*)d_in[I_NER_G1],
                                               (const float*)d_in[I_NER_BT1], NN, HD);
  packA(stream, nerh, NN, HD, nerhbf);
  gemmbf(stream, nerhbf, HD, Bnw2, out, LBL, nullptr, 0,
         (const float*)d_in[I_NER_B2], NN, LBL, 0);
}
